// LSTM_EMG_30597347017332
// MI455X (gfx1250) — matrix operand order repaired and checked by reading, not compiled, not run on hardware
//
#include <hip/hip_runtime.h>

typedef __attribute__((ext_vector_type(16))) _Float16 v16h;
typedef __attribute__((ext_vector_type(8)))  _Float16 v8h;
typedef __attribute__((ext_vector_type(8)))  float    v8f;

#define T_STEPS 128
#define BATCH   8192
#define HIDN    50
#define NIN     16
#define NCLS    20

// LDS (in halfs):
//  sB   : 3 k-chunks * 16 n-tiles * 512 halfs (WMMA B-fragment layout) = 24576
//  sWfc : 2 k-chunks * 2 n-tiles * 512 halfs                            = 2048
//  sH   : 2 buffers * 2 row-groups * 16x64 halfs                        = 4096

__device__ __forceinline__ float fsig(float x) {
    return __builtin_amdgcn_rcpf(1.0f + __expf(-x));
}
__device__ __forceinline__ float ftanh(float x) {
    return 2.0f * __builtin_amdgcn_rcpf(1.0f + __expf(-2.0f * x)) - 1.0f;
}

__global__ __launch_bounds__(256) void lstm_fused_kernel(
    const float* __restrict__ x,     // [B, T, 16]
    const float* __restrict__ W_ih,  // [200, 16]
    const float* __restrict__ W_hh,  // [200, 50]
    const float* __restrict__ b_ih,  // [200]
    const float* __restrict__ b_hh,  // [200]
    const float* __restrict__ W_fc,  // [20, 50]
    const float* __restrict__ b_fc,  // [20]
    float* __restrict__ out)         // [B, 20]
{
    __shared__ _Float16 sB[24576];
    __shared__ _Float16 sWfc[2048];
    __shared__ _Float16 sH[4096];

    const int tid  = threadIdx.x;
    const int lane = tid & 31;
    const int wave = tid >> 5;
    const int rg   = wave >> 2;        // row group (0/1): 16 batch rows each
    const int uq   = wave & 3;         // n-quarter: owns tiles {g*4+uq, g=0..3}
    const int hi16 = (lane >> 4) & 1;  // lane group
    const int ln   = lane & 15;

    // ---- pack [Whh | Wih] into WMMA B-fragment layout (f16, zero padded) ----
    // B layout (32x16 f16): lane = column n (lanes 16-31 same n), elem i -> K:
//   K = chunk*32 + (lane>=16 ? 8 : 0) + (i & 7) + (i >= 8 ? 16 : 0)
    for (int q = tid; q < 24576; q += 256) {
        int frag = q >> 9;             // 0..47
        int c    = frag >> 4;          // k-chunk 0..2
        int nt   = frag & 15;          // n-tile 0..15
        int r    = q & 511;
        int l    = r >> 4;             // lane 0..31
        int i    = r & 15;             // elem 0..15
        int K    = c * 32 + ((l & 16) ? 8 : 0) + (i & 7) + ((i & 8) ? 16 : 0);
        int n    = nt * 16 + (l & 15);
        int gate = n >> 6;
        int jj   = n & 63;
        float val = 0.0f;
        if (jj < HIDN) {
            if (c < 2) {                       // recurrent part, K = h index
                if (K < HIDN) val = W_hh[(gate * HIDN + jj) * HIDN + K];
            } else {                           // input part, K-64 = x index
                int ii = K - 64;
                if (ii < NIN) val = W_ih[(gate * HIDN + jj) * NIN + ii];
            }
        }
        sB[q] = (_Float16)val;
    }
    // ---- pack W_fc (head) into B-fragment layout ----
    for (int q = tid; q < 2048; q += 256) {
        int frag = q >> 9;             // 0..3 = c2*2 + t2
        int c2 = frag >> 1, t2 = frag & 1;
        int r = q & 511;
        int l = r >> 4, i = r & 15;
        int K = c2 * 32 + ((l & 16) ? 8 : 0) + (i & 7) + ((i & 8) ? 16 : 0);
        int n = t2 * 16 + (l & 15);
        float val = 0.0f;
        if (n < NCLS && K < HIDN) val = W_fc[n * HIDN + K];
        sWfc[q] = (_Float16)val;
    }
    // ---- zero both h double-buffers ----
    for (int q = tid; q < 4096; q += 256) sH[q] = (_Float16)0.0f;

    // persistent bias accumulators (C operand of first WMMA in each chain)
    v8f biasAcc[4];
#pragma unroll
    for (int g = 0; g < 4; ++g) {
        int n = (g * 4 + uq) * 16 + ln;
        int gate = n >> 6, jj = n & 63;
        float b = (jj < HIDN) ? (b_ih[gate * HIDN + jj] + b_hh[gate * HIDN + jj]) : 0.0f;
        biasAcc[g] = (v8f){b, b, b, b, b, b, b, b};
    }

    __syncthreads();

    // ---- hoist the 12 loop-invariant B fragments into registers ----
    v16h bfr0[4], bfr1[4], bfr2[4];
#pragma unroll
    for (int g = 0; g < 4; ++g) {
        int nt = g * 4 + uq;
        bfr0[g] = *(const v16h*)(sB + (0 * 16 + nt) * 512 + lane * 16);
        bfr1[g] = *(const v16h*)(sB + (1 * 16 + nt) * 512 + lane * 16);
        bfr2[g] = *(const v16h*)(sB + (2 * 16 + nt) * 512 + lane * 16);
    }

    const int rowBase = blockIdx.x * 32 + rg * 16;
    // x pointer for A chunk-2: lane row = rowBase + ln, j-offset by lane group
    const float* xrow = x + ((size_t)(rowBase + ln) * T_STEPS) * NIN + hi16 * 8;

    float cstate[8];
#pragma unroll
    for (int v = 0; v < 8; ++v) cstate[v] = 0.0f;

    const int kb0 = hi16 * 8;  // A-fragment interleaved K base within chunk

    // software pipeline: prefetch x(0)
    float4 xc0, xc1;
    {
        const float4* xp = (const float4*)(xrow);
        xc0 = xp[0];
        xc1 = xp[1];
    }

    for (int t = 0; t < T_STEPS; ++t) {
        // issue prefetch of x(t+1) first (clamped so it stays in-bounds);
        // consumed next iteration -> a full loop body of latency hiding
        int tn = (t + 1 < T_STEPS) ? (t + 1) : t;
        const float4* xpn = (const float4*)(xrow + (size_t)tn * NIN);
        float4 xn0 = xpn[0];
        float4 xn1 = xpn[1];

        const _Float16* hr = sH + (t & 1) * 2048 + rg * 1024;
        // A chunks 0/1 (recurrent h) from LDS; interleaved K layout:
        // lane<16: K {c*32+0..7, c*32+16..23}; lane>=16: +8
        union { v16h v; v8h h[2]; } a0, a1, a2;
        a0.h[0] = *(const v8h*)(hr + ln * 64 + kb0);
        a0.h[1] = *(const v8h*)(hr + ln * 64 + kb0 + 16);
        a1.h[0] = *(const v8h*)(hr + ln * 64 + 32 + kb0);
        a1.h[1] = *(const v8h*)(hr + ln * 64 + 32 + kb0 + 16);
        // A chunk 2 (x_t) from the previous iteration's prefetch, f32 -> f16
        {
            v8h lo = { (_Float16)xc0.x, (_Float16)xc0.y, (_Float16)xc0.z, (_Float16)xc0.w,
                       (_Float16)xc1.x, (_Float16)xc1.y, (_Float16)xc1.z, (_Float16)xc1.w };
            v8h zz = {};
            a2.h[0] = lo;
            a2.h[1] = zz;   // K 80..95 padding
        }

        // 12 WMMAs: 4 independent gate chains, chunk-major so consecutive
        // WMMAs have no D->A/B dependence and pipeline back-to-back.
        v8f acc[4];
#pragma unroll
        for (int g = 0; g < 4; ++g)
            acc[g] = __builtin_amdgcn_wmma_f32_16x16x32_f16(false, a0.v, false, bfr0[g],
                                                            (short)0, biasAcc[g], false, false);
#pragma unroll
        for (int g = 0; g < 4; ++g)
            acc[g] = __builtin_amdgcn_wmma_f32_16x16x32_f16(false, a1.v, false, bfr1[g],
                                                            (short)0, acc[g], false, false);
#pragma unroll
        for (int g = 0; g < 4; ++g)
            acc[g] = __builtin_amdgcn_wmma_f32_16x16x32_f16(false, a2.v, false, bfr2[g],
                                                            (short)0, acc[g], false, false);

        // elementwise gate combine; this wave owns cols jj = uq*16 + ln of every gate
        _Float16* hw = sH + ((t + 1) & 1) * 2048 + rg * 1024;
#pragma unroll
        for (int v = 0; v < 8; ++v) {
            float ig = fsig(acc[0][v]);
            float fg = fsig(acc[1][v]);
            float gg = ftanh(acc[2][v]);
            float og = fsig(acc[3][v]);
            float c  = fg * cstate[v] + ig * gg;
            cstate[v] = c;
            float h = og * ftanh(c);
            int m = v + hi16 * 8;                  // D row
            hw[m * 64 + uq * 16 + ln] = (_Float16)h;
        }

        xc0 = xn0;
        xc1 = xn1;
        __syncthreads();
    }

    // ---- head: logits = relu(hT) @ W_fc^T + b_fc, via 2 WMMAs ----
    // final h lives in buffer (T_STEPS & 1) == 0
    if (uq < 2) {
        const _Float16* hr = sH + rg * 1024;
        union { v16h v; v8h h[2]; } a0, a1;
        a0.h[0] = *(const v8h*)(hr + ln * 64 + kb0);
        a0.h[1] = *(const v8h*)(hr + ln * 64 + kb0 + 16);
        a1.h[0] = *(const v8h*)(hr + ln * 64 + 32 + kb0);
        a1.h[1] = *(const v8h*)(hr + ln * 64 + 32 + kb0 + 16);
#pragma unroll
        for (int i = 0; i < 16; ++i) {   // ReLU on A fragments
            float t0 = (float)a0.v[i]; a0.v[i] = (_Float16)(t0 > 0.0f ? t0 : 0.0f);
            float t1 = (float)a1.v[i]; a1.v[i] = (_Float16)(t1 > 0.0f ? t1 : 0.0f);
        }
        int t2 = uq;
        int n = t2 * 16 + ln;
        float bb = (n < NCLS) ? b_fc[n] : 0.0f;
        v8f oacc = (v8f){bb, bb, bb, bb, bb, bb, bb, bb};
        v16h bf0 = *(const v16h*)(sWfc + (0 * 2 + t2) * 512 + lane * 16);
        v16h bf1 = *(const v16h*)(sWfc + (1 * 2 + t2) * 512 + lane * 16);
        oacc = __builtin_amdgcn_wmma_f32_16x16x32_f16(false, a0.v, false, bf0,
                                                      (short)0, oacc, false, false);
        oacc = __builtin_amdgcn_wmma_f32_16x16x32_f16(false, a1.v, false, bf1,
                                                      (short)0, oacc, false, false);
        if (n < NCLS) {
#pragma unroll
            for (int v = 0; v < 8; ++v) {
                int m = v + hi16 * 8;
                out[(size_t)(rowBase + m) * NCLS + n] = oacc[v];
            }
        }
    }
}

extern "C" void kernel_launch(void* const* d_in, const int* in_sizes, int n_in,
                              void* d_out, int out_size, void* d_ws, size_t ws_size,
                              hipStream_t stream) {
    (void)in_sizes; (void)n_in; (void)d_ws; (void)ws_size; (void)out_size;
    const float* x    = (const float*)d_in[0];
    const float* W_ih = (const float*)d_in[1];
    const float* W_hh = (const float*)d_in[2];
    const float* b_ih = (const float*)d_in[3];
    const float* b_hh = (const float*)d_in[4];
    const float* W_fc = (const float*)d_in[5];
    const float* b_fc = (const float*)d_in[6];
    float* out = (float*)d_out;

    dim3 grid(BATCH / 32);   // 256 blocks, 32 batch rows each
    dim3 block(256);         // 8 waves: 2 row-groups x 4 N-quarters
    lstm_fused_kernel<<<grid, block, 0, stream>>>(x, W_ih, W_hh, b_ih, b_hh,
                                                  W_fc, b_fc, out);
}
